// LampMLP_23175643529610
// MI455X (gfx1250) — matrix operand order repaired and checked by reading, not compiled, not run on hardware
//
#include <hip/hip_runtime.h>
#include <math.h>

// ---------------------------------------------------------------------------
// LampMLP for MI455X (gfx1250): block-quantized (m_bits=3, bk=32) MLP.
// Quantized values are m * 2^(e-3), |m| <= 16 -> exactly representable in
// bf16, so both GEMMs run on v_wmma_f32_16x16x32_bf16 (K=32 == quant block).
// Double-buffered K tiles overlap global_load_b128 with WMMA issue.
// ---------------------------------------------------------------------------

typedef __attribute__((ext_vector_type(16))) __bf16 v16bf;
typedef __attribute__((ext_vector_type(8)))  __bf16 v8bf;
typedef __attribute__((ext_vector_type(8)))  float  v8f;

#define B_DIM     4
#define S_DIM     4096
#define N_EMBD    1024
#define HIDDEN    4096
#define M_ROWS    (B_DIM * S_DIM)      // 16384
#define KSTEP     32                   // WMMA K == quant block

__device__ __forceinline__ __bf16 f32_to_bf16(float f) {
    unsigned u = __float_as_uint(f);
    unsigned r = u + 0x7FFFu + ((u >> 16) & 1u);   // RNE (exact for quantized vals)
    unsigned short b = (unsigned short)(r >> 16);
    return __builtin_bit_cast(__bf16, b);
}

// scale = 2^(floor(log2(max(amax, 2^-126))) - 3), matching the reference.
__device__ __forceinline__ float block_scale_from_amax(float amax) {
    amax = fmaxf(amax, 1.17549435e-38f);                       // 2^-126 (normal)
    int e = (int)((__float_as_uint(amax) >> 23) & 0xFF) - 127; // floor(log2) exact
    return ldexpf(1.0f, e - 3);                                // handles denorm scale
}

__device__ __forceinline__ float gelu_exact(float v) {
    return 0.5f * v * (1.0f + erff(v * 0.70710678118654752f));
}

// One 32-lane wave == one quant block of 32 consecutive elements.
__global__ __launch_bounds__(256) void quant_kernel(
    const float* __restrict__ in, __bf16* __restrict__ out) {
    int i = blockIdx.x * blockDim.x + threadIdx.x;
    float v = in[i];
    float amax = fabsf(v);
#pragma unroll
    for (int off = 16; off >= 1; off >>= 1)
        amax = fmaxf(amax, __shfl_xor(amax, off, 32));
    float s = block_scale_from_amax(amax);
    float q = rintf(v / s) * s;                // round-half-even like jnp.round
    out[i] = f32_to_bf16(q);                   // exact: q has <= 5 mantissa bits
}

// One K-step worth of fragments: 2 A tiles (32 M rows) + 4 B tiles (64 N cols).
struct KTile {
    v16bf a[2];
    v16bf b[4];
};

__device__ __forceinline__ void load_ktile(
    KTile& f, const __bf16* __restrict__ arow0, const __bf16* __restrict__ arow1,
    const __bf16* __restrict__ brow0, size_t brow_stride, int k, int hf) {
    // A frag (16x32): VGPR0-3 <- K[k+hf*16 .. +7], VGPR4-7 <- K[k+hf*16+8 .. +7]
    union { v16bf v; v8bf h[2]; } af;
    af.h[0] = *(const v8bf*)(arow0 + k + hf * 16);
    af.h[1] = *(const v8bf*)(arow0 + k + hf * 16 + 8);
    f.a[0] = af.v;
    af.h[0] = *(const v8bf*)(arow1 + k + hf * 16);
    af.h[1] = *(const v8bf*)(arow1 + k + hf * 16 + 8);
    f.a[1] = af.v;
    // B frag (32x16): lane col = l16, 16 contiguous K at k (+hf*16 pre-applied)
#pragma unroll
    for (int t = 0; t < 4; ++t)
        f.b[t] = *(const v16bf*)(brow0 + (size_t)t * brow_stride + k);
}

__device__ __forceinline__ void mma_ktile(v8f acc[2][4], const KTile& f) {
#pragma unroll
    for (int m = 0; m < 2; ++m)
#pragma unroll
        for (int t = 0; t < 4; ++t)
            acc[m][t] = __builtin_amdgcn_wmma_f32_16x16x32_bf16(
                false, f.a[m], false, f.b[t], (short)0, acc[m][t], false, false);
}

// D = A[M,K] * W[N,K]^T + bias.  Wave tile: 32(M) x 64(N), WG tile: 128 x 128.
// GELU_QUANT: fused exact-GELU + block-quant(bk=32) epilogue, bf16 output.
template <bool GELU_QUANT>
__global__ __launch_bounds__(256) void gemm_kernel(
    const __bf16* __restrict__ A,    // [M,K] row-major (quantized)
    const __bf16* __restrict__ W,    // [N,K] row-major (quantized)
    const float*  __restrict__ bias, // [N]
    void* __restrict__ outp,         // bf16 [M,N] if GELU_QUANT else f32 [M,N]
    int M, int N, int K) {
    const int lane = threadIdx.x & 31;
    const int wave = threadIdx.x >> 5;
    const int hf   = lane >> 4;      // half-wave selector (CDNA5 frag layout)
    const int l16  = lane & 15;

    const int wm = wave >> 1;                         // 0..3 -> 32 M rows each
    const int wn = wave & 1;                          // 0..1 -> 64 N cols each
    const int m0 = blockIdx.y * 128 + wm * 32;
    const int nb = blockIdx.x * 128 + wn * 64;        // wave cols [nb, nb+64)

    // Accumulators initialized with bias (column-dependent only).
    v8f acc[2][4];
#pragma unroll
    for (int t = 0; t < 4; ++t) {
        float b = bias[nb + t * 16 + l16];
#pragma unroll
        for (int m = 0; m < 2; ++m)
#pragma unroll
            for (int j = 0; j < 8; ++j) acc[m][t][j] = b;
    }

    const __bf16* arow0 = A + (size_t)(m0 + l16) * K;
    const __bf16* arow1 = A + (size_t)(m0 + 16 + l16) * K;
    const __bf16* brow0 = W + (size_t)(nb + l16) * K + hf * 16;
    const size_t  bstr  = (size_t)16 * K;

    // Software-pipelined ping-pong over K (K % 64 == 0 for both GEMMs).
    KTile f0, f1;
    load_ktile(f0, arow0, arow1, brow0, bstr, 0, hf);
    for (int k = 0; k + 64 <= K; k += 64) {
        if (k + 256 < K) {  // long-range global_prefetch_b8 into near caches
            __builtin_prefetch((const void*)(arow0 + k + 256), 0, 3);
            __builtin_prefetch((const void*)(arow1 + k + 256), 0, 3);
            __builtin_prefetch((const void*)(brow0 + k + 256), 0, 3);
        }
        load_ktile(f1, arow0, arow1, brow0, bstr, k + KSTEP, hf);
        mma_ktile(acc, f0);                      // overlaps with f1 loads
        if (k + 64 < K)
            load_ktile(f0, arow0, arow1, brow0, bstr, k + 64, hf);
        mma_ktile(acc, f1);                      // overlaps with f0 loads
    }

    if (GELU_QUANT) {
        // Fused exact GELU + block_quant along N (bk=32 == two adjacent
        // 16-wide tiles of this wave; 16-lane shfl reduction = row block max).
        __bf16* out = (__bf16*)outp;
#pragma unroll
        for (int m = 0; m < 2; ++m) {
#pragma unroll
            for (int t = 0; t < 4; t += 2) {
#pragma unroll
                for (int j = 0; j < 8; ++j) {
                    float v0 = gelu_exact(acc[m][t][j]);
                    float v1 = gelu_exact(acc[m][t + 1][j]);
                    float amax = fmaxf(fabsf(v0), fabsf(v1));
#pragma unroll
                    for (int off = 8; off >= 1; off >>= 1)
                        amax = fmaxf(amax, __shfl_xor(amax, off, 32));
                    float s = block_scale_from_amax(amax);
                    float q0 = rintf(v0 / s) * s;
                    float q1 = rintf(v1 / s) * s;
                    size_t row = (size_t)(m0 + m * 16 + j + hf * 8);
                    out[row * N + nb + t * 16 + l16]       = f32_to_bf16(q0);
                    out[row * N + nb + (t + 1) * 16 + l16] = f32_to_bf16(q1);
                }
            }
        }
    } else {
        float* out = (float*)outp;
#pragma unroll
        for (int m = 0; m < 2; ++m)
#pragma unroll
            for (int t = 0; t < 4; ++t)
#pragma unroll
                for (int j = 0; j < 8; ++j)
                    out[(size_t)(m0 + m * 16 + j + hf * 8) * N + nb + t * 16 + l16] =
                        acc[m][t][j];
    }
}

extern "C" void kernel_launch(void* const* d_in, const int* in_sizes, int n_in,
                              void* d_out, int out_size, void* d_ws, size_t ws_size,
                              hipStream_t stream) {
    const float* x      = (const float*)d_in[0];  // [4,4096,1024]
    const float* w_fc   = (const float*)d_in[1];  // [4096,1024]
    const float* b_fc   = (const float*)d_in[2];  // [4096]
    const float* w_proj = (const float*)d_in[3];  // [1024,4096]
    const float* b_proj = (const float*)d_in[4];  // [1024]
    float* out = (float*)d_out;                   // [4,4096,1024] f32
    (void)in_sizes; (void)n_in; (void)out_size; (void)ws_size;

    // Workspace layout (bytes): xq 32MB | wfcq 8MB | wprq 8MB | hq 128MB
    char* ws = (char*)d_ws;
    const size_t sz_xq  = (size_t)M_ROWS * N_EMBD * 2;
    const size_t sz_wfc = (size_t)HIDDEN * N_EMBD * 2;
    const size_t sz_wpr = (size_t)N_EMBD * HIDDEN * 2;
    __bf16* xq   = (__bf16*)(ws);
    __bf16* wfcq = (__bf16*)(ws + sz_xq);
    __bf16* wprq = (__bf16*)(ws + sz_xq + sz_wfc);
    __bf16* hq   = (__bf16*)(ws + sz_xq + sz_wfc + sz_wpr);

    // 1) Block-quantize x, w_fc, w_proj to bf16 (exact representation).
    quant_kernel<<<(M_ROWS * N_EMBD) / 256, 256, 0, stream>>>(x, xq);
    quant_kernel<<<(HIDDEN * N_EMBD) / 256, 256, 0, stream>>>(w_fc, wfcq);
    quant_kernel<<<(N_EMBD * HIDDEN) / 256, 256, 0, stream>>>(w_proj, wprq);

    // 2) GEMM1 + bias + exact GELU + fused block-quant -> hq (bf16).
    dim3 g1(HIDDEN / 128, M_ROWS / 128);   // (32, 128)
    gemm_kernel<true><<<g1, 256, 0, stream>>>(xq, wfcq, b_fc, (void*)hq,
                                              M_ROWS, HIDDEN, N_EMBD);

    // 3) GEMM2 + bias -> f32 output.
    dim3 g2(N_EMBD / 128, M_ROWS / 128);   // (8, 128)
    gemm_kernel<false><<<g2, 256, 0, stream>>>(hq, wprq, b_proj, (void*)out,
                                               M_ROWS, N_EMBD, HIDDEN);
}
